// SymbolicAttentionHead_22342419874057
// MI455X (gfx1250) — matrix operand order repaired and checked by reading, not compiled, not run on hardware
//
#include <hip/hip_runtime.h>

typedef __attribute__((ext_vector_type(16))) _Float16 v16h;
typedef __attribute__((ext_vector_type(8)))  _Float16 v8h;
typedef __attribute__((ext_vector_type(4)))  unsigned v4u;
typedef __attribute__((ext_vector_type(8)))  float    v8f;
typedef int v4i_vs __attribute__((vector_size(16)));   // matches async builtin param type

#define VOCAB 1024
#define NB    8
#define NT    1024
#define ND    8192   // V*H

#define BM 128   // d per block
#define BN 128   // i per block
#define BK 32    // k per step

#if defined(__has_builtin)
#if __has_builtin(__builtin_amdgcn_global_load_async_to_lds_b128)
#define HAS_ASYNC_LDS 1
#endif
#endif

#if defined(HAS_ASYNC_LDS)
__device__ __forceinline__ void async_ld_b128(const void* g, void* l) {
  __builtin_amdgcn_global_load_async_to_lds_b128(
      (__attribute__((address_space(1))) v4i_vs*)g,
      (__attribute__((address_space(3))) v4i_vs*)l, 0, 0);
}
__device__ __forceinline__ void wait_async0() {
#if __has_builtin(__builtin_amdgcn_s_wait_asynccnt)
  __builtin_amdgcn_s_wait_asynccnt(0);
#else
  asm volatile("s_wait_asynccnt 0x0" ::: "memory");
#endif
}
#endif

// ---------------- kernel 1: value_weight f32 -> f16 table (L2-resident, 16MB) --------
__global__ void k_cvt_f16(const float* __restrict__ src, _Float16* __restrict__ dst,
                          size_t n) {
  size_t i = (size_t)blockIdx.x * blockDim.x + threadIdx.x;
  const size_t stride = (size_t)gridDim.x * blockDim.x;
  for (; i < n; i += stride) dst[i] = (_Float16)src[i];
}

// ---------------- kernel 2: att[b,i,:] = softmax(h_cooc[tok_i, tok_:]) as f16 --------
__global__ void __launch_bounds__(256) k_att(const int* __restrict__ tok,
                                             const float* __restrict__ hc,
                                             _Float16* __restrict__ att) {
  __shared__ float red[256];
  const int b = blockIdx.x >> 10;
  const int i = blockIdx.x & (NT - 1);
  const int ti = tok[b * NT + i];
  const float* __restrict__ hrow = hc + (size_t)ti * VOCAB;

  float w[4];
  float mx = -3.0e38f;
#pragma unroll
  for (int t = 0; t < 4; ++t) {
    const int j = t * 256 + threadIdx.x;
    const int tj = tok[b * NT + j];
    w[t] = hrow[tj];
    mx = fmaxf(mx, w[t]);
  }
  red[threadIdx.x] = mx;
  __syncthreads();
  for (int s = 128; s > 0; s >>= 1) {
    if ((int)threadIdx.x < s)
      red[threadIdx.x] = fmaxf(red[threadIdx.x], red[threadIdx.x + s]);
    __syncthreads();
  }
  mx = red[0];
  __syncthreads();

  float sum = 0.f;
#pragma unroll
  for (int t = 0; t < 4; ++t) { w[t] = __expf(w[t] - mx); sum += w[t]; }
  red[threadIdx.x] = sum;
  __syncthreads();
  for (int s = 128; s > 0; s >>= 1) {
    if ((int)threadIdx.x < s) red[threadIdx.x] += red[threadIdx.x + s];
    __syncthreads();
  }
  const float inv = 1.0f / red[0];

  _Float16* __restrict__ arow = att + ((size_t)b * NT + i) * NT;
#pragma unroll
  for (int t = 0; t < 4; ++t) {
    const int j = t * 256 + threadIdx.x;
    arow[j] = (_Float16)(w[t] * inv);
  }
}

// ---------------- kernel 3: LDS-tiled WMMA GEMM, out^T tiles: D = v^T x att^T --------
// Block: 128(d) x 128(i), double-buffered K-steps of 32.
// Wave: 32(d) x 64(i) = 2 A-frags x 4 B-frags = 8 WMMA per K-step.
__global__ void __launch_bounds__(256) k_gemm(const int* __restrict__ tok,
                                              const _Float16* __restrict__ vf16,
                                              const _Float16* __restrict__ att,
                                              const float* __restrict__ vw,
                                              float* __restrict__ out) {
  __shared__ unsigned rowoff[NT];            // tok[b][j] * ND
  __shared__ _Float16 As[2][BM][BK];         // v^T tile, [d][k] (transposed at store)
  __shared__ _Float16 Bs[2][BN][BK];         // att^T tile, [i][k]

  const unsigned tid  = threadIdx.x;
  const unsigned lane = tid & 31u;
  const unsigned wave = tid >> 5;
  const unsigned m    = lane & 15u;
  const unsigned hf   = lane >> 4;

  // 4096 blocks: b(8) x dblk(64) x iblk(8)
  const unsigned bx  = blockIdx.x;
  const unsigned b   = bx >> 9;
  const unsigned rem = bx & 511u;
  const unsigned d0  = (rem >> 3) * BM;
  const unsigned i0  = (rem & 7u) * BN;

  for (unsigned t = tid; t < NT; t += 256u)
    rowoff[t] = (unsigned)tok[b * NT + t] * (unsigned)ND;
  __syncthreads();

  // ---- staging assignments ----
  // A: thread handles k-pair (2kp,2kp+1) x 8 d's  (transpose via v_perm + b32 stores)
  const unsigned kp       = tid >> 4;            // 0..15
  const unsigned dseg     = (tid & 15u) * 8u;    // 0..120
  const unsigned acolBase = d0 + dseg;
  // B: thread handles one att row segment of 16 halves (32B)
  const unsigned bi   = tid >> 1;                // 0..127
  const unsigned bseg = (tid & 1u) * 16u;        // halves
  const _Float16* __restrict__ bgp =
      att + (size_t)b * NT * NT + (size_t)(i0 + bi) * NT + bseg;
  _Float16* const blp[2] = { &Bs[0][bi][bseg], &Bs[1][bi][bseg] };

  auto stageB = [&](unsigned j0n, unsigned buf) {
#if defined(HAS_ASYNC_LDS)
    const _Float16* g = bgp + j0n;
    async_ld_b128(g,     blp[buf]);
    async_ld_b128(g + 8, blp[buf] + 8);
#else
    *(v16h*)blp[buf] = *(const v16h*)(bgp + j0n);
#endif
  };
  auto loadA = [&](unsigned j0n, v4u& r0, v4u& r1) {
    const unsigned j = j0n + 2u * kp;
    r0 = *(const v4u*)(vf16 + rowoff[j]     + acolBase);   // row j,   8 d-halves
    r1 = *(const v4u*)(vf16 + rowoff[j + 1] + acolBase);   // row j+1, 8 d-halves
  };
  auto storeA = [&](unsigned buf, v4u r0, v4u r1) {
    // transpose d-major -> k-major: pack (k,k+1) half-pairs per d with v_perm_b32
#pragma unroll
    for (int w = 0; w < 4; ++w) {
      const unsigned lo = __builtin_amdgcn_perm(r1[w], r0[w], 0x05040100u);
      const unsigned hi = __builtin_amdgcn_perm(r1[w], r0[w], 0x07060302u);
      *(unsigned*)&As[buf][dseg + 2u * (unsigned)w     ][2u * kp] = lo;
      *(unsigned*)&As[buf][dseg + 2u * (unsigned)w + 1u][2u * kp] = hi;
    }
  };

  // ---- per-wave compute geometry: 32d x 64i ----
  const unsigned dd0 = (wave & 3u) * 32u;
  const unsigned ii0 = (wave >> 2) * 64u;

  v8f acc[8] = {};   // [p*4 + t], p: d-subtile, t: i-subtile

  // ---- prologue: fill buffer 0 ----
  {
    stageB(0u, 0u);
    v4u r0{}, r1{};
    loadA(0u, r0, r1);
    storeA(0u, r0, r1);
  }
#if defined(HAS_ASYNC_LDS)
  wait_async0();
#endif
  __syncthreads();

  for (unsigned jb = 0; jb < 32u; ++jb) {
    const unsigned cur  = jb & 1u;
    const unsigned nxt  = cur ^ 1u;
    const bool     more = (jb + 1u) < 32u;
    const unsigned j0n  = (jb + 1u) * BK;

    v4u r0{}, r1{};
    if (more) {
      stageB(j0n, nxt);       // async: no VGPR dependency, lands in LDS
      loadA(j0n, r0, r1);     // global loads in flight during compute
    }

    // ---- compute on buf[cur]: 2 A-frags x 4 B-frags ----
#pragma unroll
    for (int p = 0; p < 2; ++p) {
      const unsigned drow = dd0 + (unsigned)p * 16u + m;
      // A-frag: elements 0..15 -> K = 16*hf + e
      const v8h lo = *(const v8h*)&As[cur][drow][16u * hf];
      const v8h hi = *(const v8h*)&As[cur][drow][16u * hf + 8u];
      v16h af;
#pragma unroll
      for (int e = 0; e < 8; ++e) { af[e] = lo[e]; af[e + 8] = hi[e]; }
#pragma unroll
      for (int t = 0; t < 4; ++t) {
        const v16h bf = *(const v16h*)&Bs[cur][ii0 + (unsigned)t * 16u + m][16u * hf];
        acc[p * 4 + t] = __builtin_amdgcn_wmma_f32_16x16x32_f16(
            false, af, false, bf, (short)0, acc[p * 4 + t], false, false);
      }
    }

    if (more) storeA(nxt, r0, r1);
#if defined(HAS_ASYNC_LDS)
    wait_async0();
#endif
    __syncthreads();
  }

  // ---- epilogue: d contiguous per lane; add f32 identity row value_weight[tok_i] ----
#pragma unroll
  for (int p = 0; p < 2; ++p) {
    const unsigned dst0 = d0 + dd0 + (unsigned)p * 16u + hf * 8u;
#pragma unroll
    for (int t = 0; t < 4; ++t) {
      const unsigned i = i0 + ii0 + (unsigned)t * 16u + m;
      const float* __restrict__ vwrow = vw + (size_t)rowoff[i] + dst0;
      v8f r = acc[p * 4 + t];
#pragma unroll
      for (int e = 0; e < 8; ++e) r[e] += vwrow[e];
      *(v8f*)(out + ((size_t)b * NT + i) * (size_t)ND + dst0) = r;
    }
  }
}

extern "C" void kernel_launch(void* const* d_in, const int* in_sizes, int n_in,
                              void* d_out, int out_size, void* d_ws, size_t ws_size,
                              hipStream_t stream) {
  (void)in_sizes; (void)n_in; (void)out_size; (void)ws_size;
  const int*   tok = (const int*)d_in[0];
  const float* hc  = (const float*)d_in[1];
  const float* vw  = (const float*)d_in[2];
  float* out = (float*)d_out;

  _Float16* vf16 = (_Float16*)d_ws;                                    // 16 MB
  _Float16* attw = (_Float16*)((char*)d_ws +
                               (size_t)VOCAB * ND * sizeof(_Float16)); // 16 MB

  k_cvt_f16<<<4096, 256, 0, stream>>>(vw, vf16, (size_t)VOCAB * ND);
  k_att<<<NB * NT, 256, 0, stream>>>(tok, hc, attw);
  // b(8) x 64 d-blocks x 8 i-blocks = 4096 blocks of 128x128
  k_gemm<<<4096, 256, 0, stream>>>(tok, vf16, attw, vw, out);
}
